// NAPPGATVaryV1_69020124447220
// MI455X (gfx1250) — compile-verified, matrix operand order checked by reading, not run on hardware
//
#include <hip/hip_runtime.h>
#include <hip/hip_bf16.h>

#define CDIV(a,b) (((a)+(b)-1)/(b))

typedef __attribute__((ext_vector_type(16))) _Float16 v16h;
typedef __attribute__((ext_vector_type(8)))  float    v8f;

// ---------------------------------------------------------------- helpers
__device__ __forceinline__ float warpRed(float v) {
  #pragma unroll
  for (int o = 16; o > 0; o >>= 1) v += __shfl_down(v, o);
  return v;
}

__device__ __forceinline__ void atomicMaxF(float* addr, float val) {
  // ordered-int trick: works across mixed signs, init value -inf
  if (val >= 0.0f) atomicMax((int*)addr, __float_as_int(val));
  else             atomicMin((unsigned int*)addr, __float_as_uint(val));
}

// ---------------------------------------------------------------- fills / converts
__global__ void k_fill_f32(float* p, float v, int n) {
  int i = blockIdx.x * blockDim.x + threadIdx.x;
  if (i < n) p[i] = v;
}
__global__ void k_fill_i32(int* p, int v, int n) {
  int i = blockIdx.x * blockDim.x + threadIdx.x;
  if (i < n) p[i] = v;
}
__global__ void k_f32_to_f16(const float* s, _Float16* d, int n) {
  int i = blockIdx.x * blockDim.x + threadIdx.x;
  if (i < n) d[i] = (_Float16)s[i];
}

// ---------------------------------------------------------------- embedding
__global__ void k_embed(const int* x, const float* emb, float* h, int N) {
  int i = blockIdx.x * blockDim.x + threadIdx.x;
  if (i >= N * 128) return;
  int n = i >> 7, d = i & 127;
  h[i] = emb[x[n] * 128 + d];
}

// ---------------------------------------------------------------- WMMA GEMMs
// Fragment K index for (VGPR v, half hh) per CDNA5 ISA 7.12.2 (16-bit A 16x32).
__device__ __forceinline__ int frag_k(int k0, int v, int kgrp) {
  return k0 + ((v < 4) ? 0 : 16) + kgrp * 8 + 2 * (v & 3);
}

// Generic: C[M,Nc] = A[M,128](f16) x W[Nc,128]^T(f16). One wave per 16x16 tile.
__global__ void k_gemm_wmma(const _Float16* __restrict__ A,
                            const _Float16* __restrict__ W,
                            float* __restrict__ C, int M, int Nc) {
  int wave = blockIdx.x * (blockDim.x >> 5) + (threadIdx.x >> 5);
  int lane = threadIdx.x & 31;
  int tilesN = Nc >> 4;
  int nwaves = (M >> 4) * tilesN;
  if (wave >= nwaves) return;                  // wave-uniform: EXEC stays all-ones
  int tileM = wave / tilesN, tileN = wave % tilesN;
  int mrow = tileM * 16 + (lane & 15);
  int ncol = tileN * 16 + (lane & 15);
  int kgrp = lane >> 4;                        // lane group 0-15 / 16-31
  v8f c = {};
  #pragma unroll
  for (int k0 = 0; k0 < 128; k0 += 32) {
    __builtin_prefetch(&A[(size_t)mrow * 128 + k0 + 32], 0, 1);  // global_prefetch_b8
    v16h a, b;
    #pragma unroll
    for (int v = 0; v < 8; ++v) {
      int kk = frag_k(k0, v, kgrp);
      a[2*v]   = A[(size_t)mrow * 128 + kk];
      a[2*v+1] = A[(size_t)mrow * 128 + kk + 1];
      b[2*v]   = W[(size_t)ncol * 128 + kk];
      b[2*v+1] = W[(size_t)ncol * 128 + kk + 1];
    }
    c = __builtin_amdgcn_wmma_f32_16x16x32_f16(false, a, false, b,
                                               (short)0, c, false, false);
  }
  int rbase = tileM * 16 + kgrp * 8;           // C: M = r + 8*lanegrp
  #pragma unroll
  for (int r = 0; r < 8; ++r)
    C[(size_t)(rbase + r) * Nc + ncol] = c[r];
}

// Blocked: one wave computes a 64x16 output block (4 M-tiles reuse one B
// fragment -> 4x less weight traffic, 16 WMMAs per K-sweep). M % 64 == 0.
__global__ void k_gemm_wmma4(const _Float16* __restrict__ A,
                             const _Float16* __restrict__ W,
                             float* __restrict__ C, int M, int Nc) {
  __builtin_amdgcn_s_wait_tensorcnt(0);        // CDNA5 split-counter wait (no-op here)
  int wave = blockIdx.x * (blockDim.x >> 5) + (threadIdx.x >> 5);
  int lane = threadIdx.x & 31;
  int tilesN = Nc >> 4;
  int nwaves = (M >> 6) * tilesN;
  if (wave >= nwaves) return;                  // wave-uniform guard
  int blkM = wave / tilesN, tileN = wave % tilesN;
  int ncol = tileN * 16 + (lane & 15);
  int kgrp = lane >> 4;
  int mrow0 = blkM * 64 + (lane & 15);
  v8f cacc[4] = {{}, {}, {}, {}};
  #pragma unroll
  for (int k0 = 0; k0 < 128; k0 += 32) {
    v16h b;
    #pragma unroll
    for (int v = 0; v < 8; ++v) {
      int kk = frag_k(k0, v, kgrp);
      b[2*v]   = W[(size_t)ncol * 128 + kk];
      b[2*v+1] = W[(size_t)ncol * 128 + kk + 1];
    }
    #pragma unroll
    for (int t = 0; t < 4; ++t) {
      int mrow = mrow0 + t * 16;
      __builtin_prefetch(&A[(size_t)mrow * 128 + k0 + 32], 0, 1);
      v16h a;
      #pragma unroll
      for (int v = 0; v < 8; ++v) {
        int kk = frag_k(k0, v, kgrp);
        a[2*v]   = A[(size_t)mrow * 128 + kk];
        a[2*v+1] = A[(size_t)mrow * 128 + kk + 1];
      }
      cacc[t] = __builtin_amdgcn_wmma_f32_16x16x32_f16(false, a, false, b,
                                                       (short)0, cacc[t], false, false);
    }
  }
  #pragma unroll
  for (int t = 0; t < 4; ++t) {
    int rbase = blkM * 64 + t * 16 + kgrp * 8;
    #pragma unroll
    for (int r = 0; r < 8; ++r)
      C[(size_t)(rbase + r) * Nc + ncol] = cacc[t][r];
  }
}

// ---------------------------------------------------------------- GAT pieces
// ssum[n,h] = sum_d hw[n,h,d]*a_s[h,d]  (hw = projected features [N,512])
__global__ void k_node_scores(const float* hw, const float* as_, const float* ad_,
                              float* ssum, float* dsum, int N) {
  int wid = blockIdx.x * 8 + (threadIdx.x >> 5);
  int lane = threadIdx.x & 31;
  if (wid >= N * 4) return;
  int n = wid >> 2, hd = wid & 3;
  float a = 0.f, b = 0.f;
  for (int d = lane; d < 128; d += 32) {
    float hv = hw[(size_t)n * 512 + hd * 128 + d];
    a += hv * as_[hd * 128 + d];
    b += hv * ad_[hd * 128 + d];
  }
  a = warpRed(a); b = warpRed(b);
  if (lane == 0) { ssum[n * 4 + hd] = a; dsum[n * 4 + hd] = b; }
}

__global__ void k_edge_max(const int* src0, const int* dst0, const float* ssum,
                           const float* dsum, float* emax, int E, int N) {
  int i = blockIdx.x * blockDim.x + threadIdx.x;
  int ET = E + N;
  if (i >= ET * 4) return;
  int e = i >> 2, hd = i & 3;
  int s_ = (e < E) ? src0[e] : (e - E);
  int d_ = (e < E) ? dst0[e] : (e - E);
  float v = ssum[s_ * 4 + hd] + dsum[d_ * 4 + hd];
  v = (v > 0.f) ? v : 0.2f * v;                // leaky_relu(0.2)
  atomicMaxF(&emax[d_ * 4 + hd], v);
}

__global__ void k_edge_expsum(const int* src0, const int* dst0, const float* ssum,
                              const float* dsum, const float* emax,
                              float* exbuf, float* den, int E, int N) {
  int i = blockIdx.x * blockDim.x + threadIdx.x;
  int ET = E + N;
  if (i >= ET * 4) return;
  int e = i >> 2, hd = i & 3;
  int s_ = (e < E) ? src0[e] : (e - E);
  int d_ = (e < E) ? dst0[e] : (e - E);
  float v = ssum[s_ * 4 + hd] + dsum[d_ * 4 + hd];
  v = (v > 0.f) ? v : 0.2f * v;
  float ex = __expf(v - emax[d_ * 4 + hd]);
  exbuf[i] = ex;
  atomicAdd(&den[d_ * 4 + hd], ex);
}

__global__ void k_edge_agg(const int* src0, const int* dst0, const float* hw,
                           const float* exbuf, const float* den, float* acc,
                           int E, int N) {
  int wid = blockIdx.x * 8 + (threadIdx.x >> 5);
  int lane = threadIdx.x & 31;
  int ET = E + N;
  if (wid >= ET) return;
  int e = wid;
  int s_ = (e < E) ? src0[e] : (e - E);
  int d_ = (e < E) ? dst0[e] : (e - E);
  for (int j = lane; j < 512; j += 32) {
    int hd = j >> 7;
    float alpha = exbuf[(size_t)e * 4 + hd] / (den[d_ * 4 + hd] + 1e-16f);
    atomicAdd(&acc[(size_t)d_ * 512 + j], hw[(size_t)s_ * 512 + j] * alpha);
  }
}

__global__ void k_finalize(const float* acc, const float* bias, float* h, int N) {
  int i = blockIdx.x * blockDim.x + threadIdx.x;
  if (i >= N * 128) return;
  int n = i >> 7, d = i & 127;
  const float* a = acc + (size_t)n * 512;
  float v = 0.25f * (a[d] + a[128 + d] + a[256 + d] + a[384 + d]) + bias[d];
  h[i] = (v > 0.f) ? v : 0.f;                  // relu(mean over heads + bias)
}

// ---------------------------------------------------------------- SAGPooling
__global__ void k_sag_agg(const int* src0, const int* dst0, const float* h,
                          float* agg, int E) {
  int wid = blockIdx.x * 8 + (threadIdx.x >> 5);
  int lane = threadIdx.x & 31;
  if (wid >= E) return;
  int s_ = src0[wid], d_ = dst0[wid];
  for (int j = lane; j < 128; j += 32)
    atomicAdd(&agg[(size_t)d_ * 128 + j], h[(size_t)s_ * 128 + j]);
}

__global__ void k_score(const float* agg, const float* h, const float* wrel,
                        const float* brel, const float* wroot, float* score, int N) {
  int wid = blockIdx.x * 8 + (threadIdx.x >> 5);
  int lane = threadIdx.x & 31;
  if (wid >= N) return;
  float a = 0.f;
  for (int d = lane; d < 128; d += 32)
    a += agg[(size_t)wid * 128 + d] * wrel[d] + h[(size_t)wid * 128 + d] * wroot[d];
  a = warpRed(a);
  if (lane == 0) score[wid] = a + brel[0];
}

// on-device top-K: bisection on threshold + deterministic serial selection
__global__ void k_topk(const float* score, int* perm, int N, int K) {
  __shared__ float red[256];
  __shared__ float s_lo, s_hi;
  int tid = threadIdx.x;
  float mx = -3.4e38f, mn = 3.4e38f;
  for (int i = tid; i < N; i += 256) { float v = score[i]; mx = fmaxf(mx, v); mn = fminf(mn, v); }
  red[tid] = mx; __syncthreads();
  for (int s = 128; s > 0; s >>= 1) { if (tid < s) red[tid] = fmaxf(red[tid], red[tid + s]); __syncthreads(); }
  if (tid == 0) s_hi = red[0];
  __syncthreads();
  red[tid] = mn; __syncthreads();
  for (int s = 128; s > 0; s >>= 1) { if (tid < s) red[tid] = fminf(red[tid], red[tid + s]); __syncthreads(); }
  if (tid == 0) s_lo = red[0] - 1.0f;
  __syncthreads();
  for (int iter = 0; iter < 64; ++iter) {      // invariant: cnt(>lo) >= K, cnt(>hi) < K
    float mid = 0.5f * (s_lo + s_hi);
    int c = 0;
    for (int i = tid; i < N; i += 256) c += (score[i] > mid) ? 1 : 0;
    red[tid] = (float)c; __syncthreads();
    for (int s = 128; s > 0; s >>= 1) { if (tid < s) red[tid] += red[tid + s]; __syncthreads(); }
    if (tid == 0) { if ((int)red[0] >= K) s_lo = mid; else s_hi = mid; }
    __syncthreads();
  }
  if (tid == 0) {
    int cnt = 0;
    float hi = s_hi, lo = s_lo;
    for (int n = 0; n < N && cnt < K; ++n) if (score[n] > hi) perm[cnt++] = n;
    for (int n = 0; n < N && cnt < K; ++n) { float v = score[n]; if (v > lo && v <= hi) perm[cnt++] = n; }
    for (int n = 0; cnt < K; ++n) perm[cnt++] = n % N;   // unreachable safety fill
  }
}

__global__ void k_scatter(const int* perm, int* nidx, int K) {
  int k = blockIdx.x * blockDim.x + threadIdx.x;
  if (k < K) nidx[perm[k]] = k;
}

__global__ void k_xp(const int* perm, const float* h, const float* score,
                     float* xp, int K) {
  int i = blockIdx.x * blockDim.x + threadIdx.x;
  if (i >= K * 128) return;
  int k = i >> 7, d = i & 127;
  int n = perm[k];
  xp[i] = h[(size_t)n * 128 + d] * tanhf(score[n]);
}

// ---------------------------------------------------------------- DMoN
__global__ void k_softmax16(const float* slog, const float* bias, float* s, int K) {
  int k = blockIdx.x * blockDim.x + threadIdx.x;
  if (k >= K) return;
  float v[16]; float mx = -3.4e38f;
  #pragma unroll
  for (int c = 0; c < 16; ++c) { v[c] = slog[(size_t)k * 16 + c] + bias[c]; mx = fmaxf(mx, v[c]); }
  float sum = 0.f;
  #pragma unroll
  for (int c = 0; c < 16; ++c) { v[c] = __expf(v[c] - mx); sum += v[c]; }
  #pragma unroll
  for (int c = 0; c < 16; ++c) s[(size_t)k * 16 + c] = v[c] / sum;
}

// out_x[c,f] = selu( sum_k s[k,c]*xp[k,f] )
__global__ void k_outx(const float* s, const float* xp, float* out_x, int K) {
  int c = blockIdx.x;
  int f = blockIdx.y * 32 + threadIdx.x;
  float acc = 0.f;
  for (int k = 0; k < K; ++k) acc += s[(size_t)k * 16 + c] * xp[(size_t)k * 128 + f];
  const float sc = 1.0507009873554805f, al = 1.6732632423543772f;
  out_x[c * 128 + f] = (acc > 0.f) ? sc * acc : sc * al * (__expf(acc) - 1.0f);
}

// ss[c,d] = sum_k s[k,c]s[k,d]; colsum[c] = sum_k s[k,c]
__global__ void k_s_stats(const float* s, float* ss, float* colsum, int K) {
  int c = blockIdx.x, tid = threadIdx.x;
  float p[16]; float pc = 0.f;
  #pragma unroll
  for (int d = 0; d < 16; ++d) p[d] = 0.f;
  for (int k = tid; k < K; k += 256) {
    float sc_ = s[(size_t)k * 16 + c]; pc += sc_;
    #pragma unroll
    for (int d = 0; d < 16; ++d) p[d] += sc_ * s[(size_t)k * 16 + d];
  }
  __shared__ float red[256];
  for (int d = 0; d < 16; ++d) {
    red[tid] = p[d]; __syncthreads();
    for (int st = 128; st > 0; st >>= 1) { if (tid < st) red[tid] += red[tid + st]; __syncthreads(); }
    if (tid == 0) ss[c * 16 + d] = red[0];
    __syncthreads();
  }
  red[tid] = pc; __syncthreads();
  for (int st = 128; st > 0; st >>= 1) { if (tid < st) red[tid] += red[tid + st]; __syncthreads(); }
  if (tid == 0) colsum[c] = red[0];
}

__global__ void k_deg(const int* src0, const int* dst0, const int* nidx,
                      float* deg, int E) {
  int e = blockIdx.x * blockDim.x + threadIdx.x;
  if (e >= E) return;
  int ni = nidx[src0[e]], nj = nidx[dst0[e]];
  if (ni >= 0 && nj >= 0) atomicAdd(&deg[nj], 1.0f);
}

// out_adj_raw = s^T adj s, accumulated edge-wise via LDS (ds_add_f32)
__global__ void k_outadj(const int* src0, const int* dst0, const int* nidx,
                         const float* s, float* oadj, int E) {
  __shared__ float accl[256];
  int tid = threadIdx.x;
  accl[tid] = 0.f; __syncthreads();
  int stride = gridDim.x * blockDim.x;
  for (int e = blockIdx.x * blockDim.x + tid; e < E; e += stride) {
    int ni = nidx[src0[e]], nj = nidx[dst0[e]];
    if (ni >= 0 && nj >= 0) {
      #pragma unroll
      for (int c = 0; c < 16; ++c) {
        float sc_ = s[(size_t)ni * 16 + c];
        #pragma unroll
        for (int d = 0; d < 16; ++d)
          atomicAdd(&accl[c * 16 + d], sc_ * s[(size_t)nj * 16 + d]);
      }
    }
  }
  __syncthreads();
  atomicAdd(&oadj[tid], accl[tid]);
}

// losses + normalized out_adj
__global__ void k_losses(const float* oadj_raw, const float* s, const float* deg,
                         const float* ss, const float* colsum,
                         float* oadj_n, float* gp_out, int K) {
  __shared__ float red[256];
  __shared__ float s_m2, s_ca[16], s_dsv[16], s_ssF;
  int tid = threadIdx.x;
  float p = 0.f;
  for (int j = tid; j < K; j += 256) p += deg[j];
  red[tid] = p; __syncthreads();
  for (int st = 128; st > 0; st >>= 1) { if (tid < st) red[tid] += red[tid + st]; __syncthreads(); }
  if (tid == 0) s_m2 = red[0];
  __syncthreads();
  for (int c = 0; c < 16; ++c) {
    p = 0.f;
    for (int j = tid; j < K; j += 256) p += s[(size_t)j * 16 + c] * deg[j];
    red[tid] = p; __syncthreads();
    for (int st = 128; st > 0; st >>= 1) { if (tid < st) red[tid] += red[tid + st]; __syncthreads(); }
    if (tid == 0) s_ca[c] = red[0];
    __syncthreads();
  }
  float e = ss[tid];
  red[tid] = e * e; __syncthreads();
  for (int st = 128; st > 0; st >>= 1) { if (tid < st) red[tid] += red[tid + st]; __syncthreads(); }
  if (tid == 0) s_ssF = sqrtf(red[0]);
  __syncthreads();
  {
    int c = tid >> 4, d = tid & 15;
    float v = ss[tid] / s_ssF - ((c == d) ? 0.25f : 0.f);   // I/sqrt(CL), CL=16
    red[tid] = v * v; __syncthreads();
    for (int st = 128; st > 0; st >>= 1) { if (tid < st) red[tid] += red[tid + st]; __syncthreads(); }
  }
  if (tid == 0) {
    float m = s_m2 * 0.5f;
    float tr = 0.f, cc = 0.f;
    for (int c = 0; c < 16; ++c) { tr += oadj_raw[c * 16 + c]; cc += s_ca[c] * s_ca[c]; }
    float spectral = -(tr - cc / (2.f * m)) / (2.f * m);
    float ortho = sqrtf(red[0]);
    float cn = 0.f;
    for (int c = 0; c < 16; ++c) cn += colsum[c] * colsum[c];
    float cluster = sqrtf(cn) / (float)K * 4.f - 1.f;
    gp_out[0] = spectral + ortho + cluster;
  }
  __syncthreads();
  if (tid < 16) {
    float rs = 0.f;
    for (int d = 0; d < 16; ++d) if (d != tid) rs += oadj_raw[tid * 16 + d];
    s_dsv[tid] = sqrtf(rs) + 1e-15f;
  }
  __syncthreads();
  {
    int c = tid >> 4, d = tid & 15;
    float v = (c == d) ? 0.f : oadj_raw[tid];
    oadj_n[tid] = v / (s_dsv[c] * s_dsv[d]);
  }
}

// ---------------------------------------------------------------- dense tail
__global__ void k_tail(const float* out_x, const float* oadj_n,
                       const float* Wt, const float* att_s, const float* att_d,
                       const float* bt, const float* gate_w, const float* gate_b,
                       const float* trans_w, const float* trans_b,
                       const int* task, const float* task_emb,
                       const float* task_w, const float* task_b,
                       const float* fuse_w, const float* fuse_b,
                       const float* out_w, const float* out_b, float* out) {
  __shared__ float hh[16 * 512];
  __shared__ float sS[64], sD[64];
  __shared__ float al[16 * 16 * 4];
  __shared__ float dg[16 * 128];
  __shared__ float gate[16];
  __shared__ float cat[256];
  __shared__ float fuse[128];
  __shared__ float red[256];
  int tid = threadIdx.x;
  for (int idx = tid; idx < 16 * 512; idx += 256) {     // hh = out_x @ Wt^T
    int i = idx >> 9, j = idx & 511;
    float a = 0.f;
    for (int k = 0; k < 128; ++k) a += out_x[i * 128 + k] * Wt[(size_t)j * 128 + k];
    hh[idx] = a;
  }
  __syncthreads();
  if (tid < 64) {
    int j = tid >> 2, hd = tid & 3;
    float a = 0.f, b = 0.f;
    for (int d = 0; d < 128; ++d) {
      float hv = hh[j * 512 + hd * 128 + d];
      a += hv * att_s[hd * 128 + d];
      b += hv * att_d[hd * 128 + d];
    }
    sS[tid] = a; sD[tid] = b;
  }
  __syncthreads();
  for (int idx = tid; idx < 16 * 16 * 4; idx += 256) {  // al[i,j,h]
    int i = idx >> 6, j = (idx >> 2) & 15, hd = idx & 3;
    float adjv = (i == j) ? 1.f : oadj_n[i * 16 + j];
    float ev = sS[j * 4 + hd] + sD[i * 4 + hd];
    ev = (ev > 0.f) ? ev : 0.2f * ev;
    al[idx] = (adjv == 0.f) ? -3.0e38f : ev;
  }
  __syncthreads();
  if (tid < 64) {                                       // softmax over j
    int i = tid >> 2, hd = tid & 3;
    float mx = -3.4e38f;
    for (int j = 0; j < 16; ++j) mx = fmaxf(mx, al[((i << 4) + j) * 4 + hd]);
    float sum = 0.f;
    for (int j = 0; j < 16; ++j) { float w = __expf(al[((i << 4) + j) * 4 + hd] - mx); al[((i << 4) + j) * 4 + hd] = w; sum += w; }
    for (int j = 0; j < 16; ++j) al[((i << 4) + j) * 4 + hd] /= sum;
  }
  __syncthreads();
  for (int idx = tid; idx < 16 * 128; idx += 256) {     // dg = relu(mean_h att @ hh + bt)
    int i = idx >> 7, d = idx & 127;
    float a = 0.f;
    for (int hd = 0; hd < 4; ++hd)
      for (int j = 0; j < 16; ++j)
        a += al[((i << 4) + j) * 4 + hd] * hh[j * 512 + hd * 128 + d];
    a = a * 0.25f + bt[d];
    dg[idx] = (a > 0.f) ? a : 0.f;
  }
  __syncthreads();
  if (tid < 16) {
    float a = 0.f;
    for (int d = 0; d < 128; ++d) a += dg[tid * 128 + d] * gate_w[d];
    gate[tid] = a + gate_b[0];
  }
  __syncthreads();
  if (tid == 0) {
    float mx = -3.4e38f;
    for (int i = 0; i < 16; ++i) mx = fmaxf(mx, gate[i]);
    float sum = 0.f;
    for (int i = 0; i < 16; ++i) { gate[i] = __expf(gate[i] - mx); sum += gate[i]; }
    for (int i = 0; i < 16; ++i) gate[i] /= sum;
  }
  __syncthreads();
  if (tid < 128) {                                      // ro
    float a = 0.f;
    for (int i = 0; i < 16; ++i) {
      float t = 0.f;
      for (int e = 0; e < 128; ++e) t += dg[i * 128 + e] * trans_w[(size_t)tid * 128 + e];
      a += gate[i] * (t + trans_b[tid]);
    }
    cat[tid] = a;
  }
  __syncthreads();
  if (tid < 128) {                                      // te
    float a = 0.f;
    int tk = task[0];
    for (int e = 0; e < 128; ++e) a += task_emb[(size_t)tk * 128 + e] * task_w[(size_t)tid * 128 + e];
    a += task_b[tid];
    cat[128 + tid] = (a > 0.f) ? a : 0.f;
  }
  __syncthreads();
  if (tid < 128) {                                      // fuse
    float a = 0.f;
    for (int e = 0; e < 256; ++e) a += cat[e] * fuse_w[(size_t)tid * 256 + e];
    a += fuse_b[tid];
    fuse[tid] = (a > 0.f) ? a : 0.f;
  }
  __syncthreads();
  red[tid] = (tid < 128) ? fuse[tid] * out_w[tid] : 0.f;
  __syncthreads();
  for (int st = 128; st > 0; st >>= 1) { if (tid < st) red[tid] += red[tid + st]; __syncthreads(); }
  if (tid == 0) out[0] = red[0] + out_b[0];
}

// ---------------------------------------------------------------- launch
extern "C" void kernel_launch(void* const* d_in, const int* in_sizes, int n_in,
                              void* d_out, int out_size, void* d_ws, size_t ws_size,
                              hipStream_t stream) {
  (void)in_sizes; (void)n_in; (void)out_size; (void)ws_size;
  const int N = 16000, E = 256000, K = 4000;
  const int ET = E + N;

  const int*   x        = (const int*)d_in[0];
  const int*   ei       = (const int*)d_in[1];
  const int*   src0     = ei;
  const int*   dst0     = ei + E;
  const int*   task     = (const int*)d_in[3];
  const float* node_emb = (const float*)d_in[4];
  const float* task_emb = (const float*)d_in[5];
  const float* Wl[3]  = {(const float*)d_in[6],  (const float*)d_in[10], (const float*)d_in[14]};
  const float* ASl[3] = {(const float*)d_in[7],  (const float*)d_in[11], (const float*)d_in[15]};
  const float* ADl[3] = {(const float*)d_in[8],  (const float*)d_in[12], (const float*)d_in[16]};
  const float* Bl[3]  = {(const float*)d_in[9],  (const float*)d_in[13], (const float*)d_in[17]};
  const float* sag_wrel  = (const float*)d_in[18];
  const float* sag_brel  = (const float*)d_in[19];
  const float* sag_wroot = (const float*)d_in[20];
  const float* dmon_w    = (const float*)d_in[21];
  const float* dmon_b    = (const float*)d_in[22];
  const float* Wt     = (const float*)d_in[23];
  const float* att_s  = (const float*)d_in[24];
  const float* att_d  = (const float*)d_in[25];
  const float* bt     = (const float*)d_in[26];
  const float* gate_w = (const float*)d_in[27];
  const float* gate_b = (const float*)d_in[28];
  const float* trans_w = (const float*)d_in[29];
  const float* trans_b = (const float*)d_in[30];
  const float* task_w  = (const float*)d_in[31];
  const float* task_b  = (const float*)d_in[32];
  const float* fuse_w  = (const float*)d_in[33];
  const float* fuse_b  = (const float*)d_in[34];
  const float* out_w   = (const float*)d_in[35];
  const float* out_b   = (const float*)d_in[36];
  float* out = (float*)d_out;

  // workspace carve (~87 MB)
  char* ws = (char*)d_ws;
  size_t off = 0;
  auto carve = [&](size_t bytes) -> char* {
    char* p = ws + off;
    off = (off + bytes + 255) & ~(size_t)255;
    return p;
  };
  float*    h     = (float*)carve((size_t)N * 128 * 4);
  float*    hw    = (float*)carve((size_t)N * 512 * 4);
  float*    acc   = (float*)carve((size_t)N * 512 * 4);
  _Float16* h16   = (_Float16*)carve((size_t)N * 128 * 2);
  _Float16* w16   = (_Float16*)carve((size_t)512 * 128 * 2);
  float*    ssum  = (float*)carve((size_t)N * 4 * 4);
  float*    dsum  = (float*)carve((size_t)N * 4 * 4);
  float*    emax  = (float*)carve((size_t)N * 4 * 4);
  float*    den   = (float*)carve((size_t)N * 4 * 4);
  float*    exbuf = (float*)carve((size_t)ET * 4 * 4);
  float*    score = (float*)carve((size_t)N * 4);
  int*      perm  = (int*)carve((size_t)K * 4);
  int*      nidx  = (int*)carve((size_t)N * 4);
  float*    xp    = (float*)carve((size_t)K * 128 * 4);
  _Float16* xp16  = (_Float16*)carve((size_t)K * 128 * 2);
  _Float16* dmw16 = (_Float16*)carve((size_t)16 * 128 * 2);
  float*    slog  = (float*)carve((size_t)K * 16 * 4);
  float*    smat  = (float*)carve((size_t)K * 16 * 4);
  float*    outx  = (float*)carve((size_t)16 * 128 * 4);
  float*    ssm   = (float*)carve(256 * 4);
  float*    colsum = (float*)carve(16 * 4);
  float*    deg   = (float*)carve((size_t)K * 4);
  float*    oadj  = (float*)carve(256 * 4);
  float*    oadjn = (float*)carve(256 * 4);

  dim3 blk(256);
  k_embed<<<CDIV(N * 128, 256), blk, 0, stream>>>(x, node_emb, h, N);

  for (int L = 0; L < 3; ++L) {
    k_f32_to_f16<<<CDIV(N * 128, 256), blk, 0, stream>>>(h, h16, N * 128);
    k_f32_to_f16<<<CDIV(512 * 128, 256), blk, 0, stream>>>(Wl[L], w16, 512 * 128);
    int waves = (N / 64) * (512 / 16);         // blocked GEMM: 64 rows per wave
    k_gemm_wmma4<<<CDIV(waves, 8), blk, 0, stream>>>(h16, w16, hw, N, 512);
    k_node_scores<<<CDIV(N * 4, 8), blk, 0, stream>>>(hw, ASl[L], ADl[L], ssum, dsum, N);
    k_fill_f32<<<CDIV(N * 4, 256), blk, 0, stream>>>(emax, -3.4e38f, N * 4);
    k_fill_f32<<<CDIV(N * 4, 256), blk, 0, stream>>>(den, 0.f, N * 4);
    k_fill_f32<<<CDIV(N * 512, 256), blk, 0, stream>>>(acc, 0.f, N * 512);
    k_edge_max<<<CDIV(ET * 4, 256), blk, 0, stream>>>(src0, dst0, ssum, dsum, emax, E, N);
    k_edge_expsum<<<CDIV(ET * 4, 256), blk, 0, stream>>>(src0, dst0, ssum, dsum, emax, exbuf, den, E, N);
    k_edge_agg<<<CDIV(ET, 8), blk, 0, stream>>>(src0, dst0, hw, exbuf, den, acc, E, N);
    k_finalize<<<CDIV(N * 128, 256), blk, 0, stream>>>(acc, Bl[L], h, N);
  }

  // SAGPooling
  k_fill_f32<<<CDIV(N * 128, 256), blk, 0, stream>>>(acc, 0.f, N * 128);
  k_sag_agg<<<CDIV(E, 8), blk, 0, stream>>>(src0, dst0, h, acc, E);
  k_score<<<CDIV(N, 8), blk, 0, stream>>>(acc, h, sag_wrel, sag_brel, sag_wroot, score, N);
  k_topk<<<1, blk, 0, stream>>>(score, perm, N, K);
  k_fill_i32<<<CDIV(N, 256), blk, 0, stream>>>(nidx, -1, N);
  k_scatter<<<CDIV(K, 256), blk, 0, stream>>>(perm, nidx, K);
  k_xp<<<CDIV(K * 128, 256), blk, 0, stream>>>(perm, h, score, xp, K);

  // DMoN (M=4000 not a multiple of 64 -> generic 1-tile WMMA kernel)
  k_f32_to_f16<<<CDIV(K * 128, 256), blk, 0, stream>>>(xp, xp16, K * 128);
  k_f32_to_f16<<<CDIV(16 * 128, 256), blk, 0, stream>>>(dmon_w, dmw16, 16 * 128);
  {
    int waves = (K / 16) * (16 / 16);
    k_gemm_wmma<<<CDIV(waves, 8), blk, 0, stream>>>(xp16, dmw16, slog, K, 16);
  }
  k_softmax16<<<CDIV(K, 256), blk, 0, stream>>>(slog, dmon_b, smat, K);
  k_outx<<<dim3(16, 4), dim3(32), 0, stream>>>(smat, xp, outx, K);
  k_s_stats<<<16, blk, 0, stream>>>(smat, ssm, colsum, K);
  k_fill_f32<<<CDIV(K, 256), blk, 0, stream>>>(deg, 0.f, K);
  k_deg<<<CDIV(E, 256), blk, 0, stream>>>(src0, dst0, nidx, deg, E);
  k_fill_f32<<<1, blk, 0, stream>>>(oadj, 0.f, 256);
  k_outadj<<<256, blk, 0, stream>>>(src0, dst0, nidx, smat, oadj, E);
  k_losses<<<1, blk, 0, stream>>>(oadj, smat, deg, ssm, colsum, oadjn, out + 1, K);
  k_tail<<<1, blk, 0, stream>>>(outx, oadjn, Wt, att_s, att_d, bt, gate_w, gate_b,
                                trans_w, trans_b, task, task_emb, task_w, task_b,
                                fuse_w, fuse_b, out_w, out_b, out);
}
